// GCNConv_28613072126264
// MI455X (gfx1250) — matrix operand order repaired and checked by reading, not compiled, not run on hardware
//
#include <hip/hip_runtime.h>
#include <hip/hip_bf16.h>
#include <math.h>

typedef __attribute__((ext_vector_type(16))) _Float16 v16h;
typedef __attribute__((ext_vector_type(8)))  float    v8f;
typedef __attribute__((ext_vector_type(4)))  float    f32x4;

#define NN    8192
#define DD    128
#define KSPLIT 4
#define KPER  (NN / KSPLIT)   // 2048

union Bfrag { v16h h; f32x4 f[2]; };

// Non-temporal streaming load of a float4 (A is 256MB, read-once: keep it
// out of the L2 working set so ht / partials stay resident).
__device__ __forceinline__ f32x4 ld_nt(const float* p) {
    return __builtin_nontemporal_load((const f32x4*)p);
}

// ---------------------------------------------------------------------------
// Kernel 1: d[i] = rsqrt(rowsum(A[i,:]) - A[i,i] + 1)
// ---------------------------------------------------------------------------
__global__ __launch_bounds__(256) void deg_kernel(const float* __restrict__ A,
                                                  float* __restrict__ d) {
    __shared__ float red[256];
    const int i = blockIdx.x;
    const float* row = A + (size_t)i * NN;
    float s = 0.0f;
    for (int c = threadIdx.x * 4; c < NN; c += 256 * 4) {
        f32x4 v = ld_nt(row + c);
        s += v[0] + v[1] + v[2] + v[3];
    }
    red[threadIdx.x] = s;
    __syncthreads();
    for (int off = 128; off > 0; off >>= 1) {
        if (threadIdx.x < off) red[threadIdx.x] += red[threadIdx.x + off];
        __syncthreads();
    }
    if (threadIdx.x == 0) {
        float deg = red[0] - row[i] + 1.0f;   // force unit diagonal
        d[i] = rsqrtf(deg);
    }
}

// ---------------------------------------------------------------------------
// Kernel 2: ht[c][j] = f16( d[j] * (x @ W + b)[j][c] ), ht is 128 x 8192 (transposed)
// ---------------------------------------------------------------------------
__global__ __launch_bounds__(256) void h_kernel(const float* __restrict__ x,
                                                const float* __restrict__ W,
                                                const float* __restrict__ bias,
                                                const float* __restrict__ d,
                                                _Float16* __restrict__ ht) {
    __shared__ float hl[32][DD + 1];
    const int j0   = blockIdx.x * 32;
    const int c    = threadIdx.x & (DD - 1);
    const int half = threadIdx.x >> 7;   // 0 or 1
    for (int p = 0; p < 16; ++p) {
        const int jl = p * 2 + half;
        const int j  = j0 + jl;
        const float* xr = x + (size_t)j * DD;
        float acc = bias[c];
        #pragma unroll 8
        for (int k = 0; k < DD; ++k) acc += xr[k] * W[k * DD + c];
        hl[jl][c] = acc;
    }
    __syncthreads();
    for (int idx = threadIdx.x; idx < 32 * DD; idx += 256) {
        const int cc = idx >> 5;        // 0..127
        const int jj = idx & 31;        // 0..31
        ht[(size_t)cc * NN + j0 + jj] = (_Float16)(hl[jj][cc] * d[j0 + jj]);
    }
}

// ---------------------------------------------------------------------------
// Kernel 3: GEMM  part[ks][i][c] = sum_{k in split ks} A[i][k] * ht[c][k]
// Workgroup: 8 waves, 128 rows; wave w -> rows m0+16w, all 128 cols.
// Software pipeline: B loads for step k issued before the 8 WMMAs; A floats
// for step k+32 prefetched (non-temporal) under the WMMA block.
// ---------------------------------------------------------------------------
__global__ __launch_bounds__(256) void mm_kernel(const float* __restrict__ A,
                                                 const _Float16* __restrict__ ht,
                                                 float* __restrict__ part) {
    const int wave = threadIdx.x >> 5;
    const int lane = threadIdx.x & 31;
    const int r    = lane & 15;           // row within tile (A) / col within tile (B)
    const int hi   = lane >> 4;           // lane half selector
    const int mblk = blockIdx.x & 63;     // 64 row-blocks of 128
    const int ks   = blockIdx.x >> 6;     // K split 0..3

    const int arow = mblk * 128 + wave * 16 + r;
    const float* Arow = A + (size_t)arow * NN;
    const int kbeg = ks * KPER;
    const int kend = kbeg + KPER;

    v8f acc[8] = {};

    // ---- preload A floats for the first k-step (streaming / TH_NT) ----
    const float* ap = Arow + kbeg + hi * 8;
    f32x4 f0 = ld_nt(ap + 0);
    f32x4 f1 = ld_nt(ap + 4);
    f32x4 f2 = ld_nt(ap + 16);
    f32x4 f3 = ld_nt(ap + 20);

    for (int k0 = kbeg; k0 < kend; k0 += 32) {
        // ---- convert prefetched A floats into the 16x32 f16 fragment ----
        //   hi=0: v0-3 <- K k0..k0+7      v4-7 <- K k0+16..k0+23
        //   hi=1: v0-3 <- K k0+8..k0+15   v4-7 <- K k0+24..k0+31
        v16h a;
        a[0]  = (_Float16)f0[0]; a[1]  = (_Float16)f0[1];
        a[2]  = (_Float16)f0[2]; a[3]  = (_Float16)f0[3];
        a[4]  = (_Float16)f1[0]; a[5]  = (_Float16)f1[1];
        a[6]  = (_Float16)f1[2]; a[7]  = (_Float16)f1[3];
        a[8]  = (_Float16)f2[0]; a[9]  = (_Float16)f2[1];
        a[10] = (_Float16)f2[2]; a[11] = (_Float16)f2[3];
        a[12] = (_Float16)f3[0]; a[13] = (_Float16)f3[1];
        a[14] = (_Float16)f3[2]; a[15] = (_Float16)f3[3];

        // ---- issue ALL B-fragment loads for this k-step (16 x b128) ----
        // lanes 0-15 hold K k0..k0+7,k0+16..k0+23; lanes 16-31 hold K k0+8..k0+15,k0+24..k0+31
        Bfrag b[8];
        #pragma unroll
        for (int nt = 0; nt < 8; ++nt) {
            const _Float16* bp = ht + (size_t)(nt * 16 + r) * NN + k0 + hi * 8;
            b[nt].f[0] = *(const f32x4*)(bp + 0);
            b[nt].f[1] = *(const f32x4*)(bp + 16);
        }

        // ---- prefetch next k-step's A floats (clamped; wave-uniform select) ----
        const int kn = (k0 + 32 < kend) ? (k0 + 32) : kbeg;
        const float* an = Arow + kn + hi * 8;
        f0 = ld_nt(an + 0);
        f1 = ld_nt(an + 4);
        f2 = ld_nt(an + 16);
        f3 = ld_nt(an + 20);

        // ---- 8 WMMAs; waits on b[nt] become progressive ----
        #pragma unroll
        for (int nt = 0; nt < 8; ++nt) {
            acc[nt] = __builtin_amdgcn_wmma_f32_16x16x32_f16(
                false, a, false, b[nt].h, (short)0, acc[nt], false, false);
        }
    }

    // ---- store partials: acc VGPR v -> (M = v + 8*hi, N = r) ----
    float* pout = part + (size_t)ks * NN * DD;
    const int row0 = mblk * 128 + wave * 16 + hi * 8;
    #pragma unroll
    for (int nt = 0; nt < 8; ++nt) {
        const int col = nt * 16 + r;
        #pragma unroll
        for (int v = 0; v < 8; ++v) {
            pout[(size_t)(row0 + v) * DD + col] = acc[nt][v];
        }
    }
}

// ---------------------------------------------------------------------------
// Kernel 4: out[i][c] = d[i] * sum_ks part[ks][i][c]   (deterministic order)
// ---------------------------------------------------------------------------
__global__ __launch_bounds__(256) void final_kernel(const float* __restrict__ part,
                                                    const float* __restrict__ d,
                                                    float* __restrict__ out) {
    const int i = blockIdx.x * 256 + threadIdx.x;   // over NN*DD = 1M
    const int row = i >> 7;
    float s = part[i];
    s += part[i + 1 * NN * DD];
    s += part[i + 2 * NN * DD];
    s += part[i + 3 * NN * DD];
    out[i] = d[row] * s;
}

// ---------------------------------------------------------------------------
extern "C" void kernel_launch(void* const* d_in, const int* in_sizes, int n_in,
                              void* d_out, int out_size, void* d_ws, size_t ws_size,
                              hipStream_t stream) {
    const float* A    = (const float*)d_in[0];   // 8192 x 8192
    const float* x    = (const float*)d_in[1];   // 8192 x 128
    const float* W    = (const float*)d_in[2];   // 128 x 128
    const float* bias = (const float*)d_in[3];   // 128
    float* out = (float*)d_out;                  // 8192 x 128

    // workspace layout: d (32KB) | ht f16 (2MB) | partials fp32 (4 x 4MB)
    char* ws = (char*)d_ws;
    float*    d_deg = (float*)ws;
    _Float16* ht    = (_Float16*)(ws + NN * sizeof(float));
    float*    part  = (float*)(ws + NN * sizeof(float) + (size_t)DD * NN * sizeof(_Float16));

    deg_kernel  <<<NN, 256, 0, stream>>>(A, d_deg);
    h_kernel    <<<NN / 32, 256, 0, stream>>>(x, W, bias, d_deg, ht);
    mm_kernel   <<<64 * KSPLIT, 256, 0, stream>>>(A, ht, part);
    final_kernel<<<(NN * DD) / 256, 256, 0, stream>>>(part, d_deg, out);
}
